// DiTBlock_81570018886315
// MI455X (gfx1250) — matrix operand order repaired and checked by reading, not compiled, not run on hardware
//
#include <hip/hip_runtime.h>

// ---------------------------------------------------------------------------
// DiT block for MI455X (gfx1250, wave32, WMMA).
// All GEMMs + attention matmuls use v_wmma_f32_16x16x32_bf16 (f32 accum).
// GEMM tile staging uses GLOBAL_LOAD_ASYNC_TO_LDS_B128 (ASYNCcnt) when the
// builtin is available, else synchronous load+ds_store fallback.
// ---------------------------------------------------------------------------

typedef __attribute__((ext_vector_type(16))) __bf16       bf16x16;
typedef __attribute__((ext_vector_type(8)))  float        f32x8;
typedef __attribute__((ext_vector_type(4)))  unsigned int u32x4;
typedef int v4i_gcc __attribute__((__vector_size__(4 * sizeof(int))));

union Frag { bf16x16 v; u32x4 q[2]; };

__device__ __forceinline__ f32x8 wmma_bf16(const Frag& a, const Frag& b, f32x8 c) {
  // (neg_a, A, neg_b, B, c_mod, C, reuse_a, reuse_b)
  return __builtin_amdgcn_wmma_f32_16x16x32_bf16(false, a.v, false, b.v,
                                                 (short)0, c, false, false);
}

#if defined(__has_builtin)
#  if __has_builtin(__builtin_amdgcn_global_load_async_to_lds_b128)
#    define USE_ASYNC_LDS 1
#  endif
#endif
#ifndef USE_ASYNC_LDS
#  define USE_ASYNC_LDS 0
#endif

// exact parameter types per hipcc diagnostic: AS(1)/AS(3) pointers to v4i
#define AS1P128(p) ((__attribute__((address_space(1))) v4i_gcc*)(p))
#define AS3P128(p) ((__attribute__((address_space(3))) v4i_gcc*)(p))

__device__ __forceinline__ void wait_async0() {
#if defined(__has_builtin) && __has_builtin(__builtin_amdgcn_s_wait_asynccnt)
  __builtin_amdgcn_s_wait_asynccnt(0);
#else
  asm volatile("s_wait_asynccnt 0" ::: "memory");
#endif
}

#define DH   1024      // hidden dim
#define BSZ  4
#define SEQ  1024
#define NTOK (BSZ * SEQ)   // 4096
#define NHEAD 16
#define HDIM  64

// ---------------------------------------------------------------------------
// f32 -> bf16 conversion
// ---------------------------------------------------------------------------
__global__ void __launch_bounds__(256) f32_to_bf16(const float* __restrict__ in,
                                                   __bf16* __restrict__ out, int n) {
  int i = blockIdx.x * 256 + threadIdx.x;
  if (i < n) out[i] = (__bf16)in[i];
}

// ---------------------------------------------------------------------------
// adaLN GEMV: out[bi][o] = sum_k temb[bi][k] * w[o][k] + bias[o]
// grid (4, BSZ), block 256
// ---------------------------------------------------------------------------
__global__ void __launch_bounds__(256) ada_gemv(const float* __restrict__ temb,
                                                const float* __restrict__ w,
                                                const float* __restrict__ bias,
                                                float* __restrict__ out) {
  int o  = blockIdx.x * 256 + threadIdx.x;   // 0..1023
  int bi = blockIdx.y;
  const float* t  = temb + bi * DH;
  const float* wr = w + (size_t)o * DH;
  float acc = 0.f;
#pragma unroll 4
  for (int k = 0; k < DH; ++k) acc += t[k] * wr[k];
  out[bi * DH + o] = acc + bias[o];
}

// ---------------------------------------------------------------------------
// Fused LayerNorm + adaLN modulation, f32 in -> bf16 out.
// grid NTOK, block 256 (each thread handles 4 columns)
// ---------------------------------------------------------------------------
__global__ void __launch_bounds__(256) ln_ada(const float* __restrict__ x,
                                              const float* __restrict__ g,
                                              const float* __restrict__ bta,
                                              const float* __restrict__ scale,
                                              const float* __restrict__ offset,
                                              __bf16* __restrict__ out) {
  __shared__ float sred[16];
  const int row = blockIdx.x;
  const int tid = threadIdx.x;
  const float* xr = x + (size_t)row * DH;
  float xv[4], s = 0.f, s2 = 0.f;
#pragma unroll
  for (int j = 0; j < 4; ++j) {
    xv[j] = xr[tid + j * 256];
    s += xv[j]; s2 += xv[j] * xv[j];
  }
#pragma unroll
  for (int m = 16; m >= 1; m >>= 1) { s += __shfl_xor(s, m); s2 += __shfl_xor(s2, m); }
  int wid = tid >> 5;
  if ((tid & 31) == 0) { sred[wid] = s; sred[8 + wid] = s2; }
  __syncthreads();
  if (tid == 0) {
    float a = 0.f, b = 0.f;
    for (int w = 0; w < 8; ++w) { a += sred[w]; b += sred[8 + w]; }
    sred[0] = a; sred[8] = b;
  }
  __syncthreads();
  float mean = sred[0] * (1.f / DH);
  float var  = sred[8] * (1.f / DH) - mean * mean;
  float inv  = rsqrtf(var + 1e-5f);
  int bi = row >> 10;
#pragma unroll
  for (int j = 0; j < 4; ++j) {
    int c = tid + j * 256;
    float h  = (xv[j] - mean) * inv * g[c] + bta[c];
    float sc = scale[bi * DH + c], of = offset[bi * DH + c];
    out[(size_t)row * DH + c] = (__bf16)((1.f + sc) * h + of);
  }
}

// ---------------------------------------------------------------------------
// WMMA GEMM: C[M,N] = A[M,K] @ W[N,K]^T + bias  [+resid] [GELU] -> f32 or bf16
// block 256 (8 waves), block tile 128x64, K-step 32.
// ---------------------------------------------------------------------------
template<bool RESID, bool GELU_E, bool OUTBF>
__global__ void __launch_bounds__(256)
gemm_wmma(const __bf16* __restrict__ A, const __bf16* __restrict__ W,
          const float* __restrict__ bias, const float* __restrict__ resid,
          float* __restrict__ outF, __bf16* __restrict__ outB,
          int M, int N, int K) {
  __shared__ __attribute__((aligned(16))) __bf16 As[128][40];  // 40*2B row = 80B (16B mult)
  __shared__ __attribute__((aligned(16))) __bf16 Bs[64][40];
  const int tid    = threadIdx.x;
  const int lane   = tid & 31;
  const int wid    = tid >> 5;
  const int wm     = wid & 3;      // 4 waves along M
  const int wn     = wid >> 2;     // 2 waves along N
  const int lane16 = lane & 15;
  const int hsel   = (lane < 16) ? 0 : 1;
  const int m0 = blockIdx.y * 128;
  const int n0 = blockIdx.x * 64;

  const f32x8 z = {0.f,0.f,0.f,0.f,0.f,0.f,0.f,0.f};
  f32x8 acc[2][2];
  acc[0][0] = z; acc[0][1] = z; acc[1][0] = z; acc[1][1] = z;

  const int ra0 = tid >> 2,          kca = (tid & 3) << 3;   // A chunk 0
  const int ra1 = (tid + 256) >> 2;                          // A chunk 1 (same kca)
  const int rb  = tid >> 2,          kcb = (tid & 3) << 3;   // B chunk

  for (int k0 = 0; k0 < K; k0 += 32) {
    // cooperative stage: A tile 128x32, W tile 64x32 (both bf16)
#if USE_ASYNC_LDS
    // ASYNCcnt-tracked DMA: global -> LDS, no VGPR round-trip
    __builtin_amdgcn_global_load_async_to_lds_b128(
        AS1P128(A + (size_t)(m0 + ra0) * K + k0 + kca), AS3P128(&As[ra0][kca]), 0, 0);
    __builtin_amdgcn_global_load_async_to_lds_b128(
        AS1P128(A + (size_t)(m0 + ra1) * K + k0 + kca), AS3P128(&As[ra1][kca]), 0, 0);
    __builtin_amdgcn_global_load_async_to_lds_b128(
        AS1P128(W + (size_t)(n0 + rb) * K + k0 + kcb), AS3P128(&Bs[rb][kcb]), 0, 0);
#else
    *(u32x4*)&As[ra0][kca] = *(const u32x4*)(A + (size_t)(m0 + ra0) * K + k0 + kca);
    *(u32x4*)&As[ra1][kca] = *(const u32x4*)(A + (size_t)(m0 + ra1) * K + k0 + kca);
    *(u32x4*)&Bs[rb][kcb]  = *(const u32x4*)(W + (size_t)(n0 + rb) * K + k0 + kcb);
#endif
    if (k0 + 32 < K) {   // hint next K-slab into cache (global_prefetch_b8)
      __builtin_prefetch(A + (size_t)(m0 + ra0) * K + k0 + 32 + kca, 0, 1);
      __builtin_prefetch(W + (size_t)(n0 + rb) * K + k0 + 32 + kcb, 0, 1);
    }
#if USE_ASYNC_LDS
    wait_async0();
#endif
    __syncthreads();

    Frag a[2], b[2];
    const int kba = hsel ? 8 : 0;
#pragma unroll
    for (int i = 0; i < 2; ++i) {
      int r = wm * 32 + i * 16 + lane16;
      a[i].q[0] = *(const u32x4*)&As[r][kba];        // K 0..7  / 8..15
      a[i].q[1] = *(const u32x4*)&As[r][kba + 16];   // K 16..23 / 24..31
    }
    const int kbb = hsel ? 8 : 0;
#pragma unroll
    for (int j = 0; j < 2; ++j) {
      int c = wn * 32 + j * 16 + lane16;
      b[j].q[0] = *(const u32x4*)&Bs[c][kbb];        // K 0..7  / 8..15
      b[j].q[1] = *(const u32x4*)&Bs[c][kbb + 16];   // K 16..23 / 24..31
    }
#pragma unroll
    for (int i = 0; i < 2; ++i)
#pragma unroll
      for (int j = 0; j < 2; ++j)
        acc[i][j] = wmma_bf16(a[i], b[j], acc[i][j]);
    __syncthreads();
  }

#pragma unroll
  for (int i = 0; i < 2; ++i) {
#pragma unroll
    for (int j = 0; j < 2; ++j) {
      int mb = m0 + wm * 32 + i * 16 + hsel * 8;
      int nb = n0 + wn * 32 + j * 16 + lane16;
      float bv = bias[nb];
#pragma unroll
      for (int c = 0; c < 8; ++c) {
        int m = mb + c;
        float v = acc[i][j][c] + bv;
        if (RESID)  v += resid[(size_t)m * N + nb];
        if (GELU_E) v = 0.5f * v * (1.0f + erff(v * 0.70710678118654752f));
        if (OUTBF)  outB[(size_t)m * N + nb] = (__bf16)v;
        else        outF[(size_t)m * N + nb] = v;
      }
    }
  }
}

// ---------------------------------------------------------------------------
// V transpose: qkv[b,n, 2048 + h*64+dh] -> vT[((b*16+h)*64+dh)*1024 + n]
// ---------------------------------------------------------------------------
__global__ void __launch_bounds__(256) v_transpose(const __bf16* __restrict__ qkv,
                                                   __bf16* __restrict__ vT) {
  int idx = blockIdx.x * 256 + threadIdx.x;       // 0 .. 4M-1
  int n  = idx & 1023;
  int r  = idx >> 10;
  int dh = r & 63;  r >>= 6;
  int h  = r & 15;  int b = r >> 4;
  vT[idx] = qkv[(size_t)(b * SEQ + n) * 3072 + 2048 + h * HDIM + dh];
}

// ---------------------------------------------------------------------------
// Flash attention (streaming softmax), bf16 WMMA, f32 accum.
// grid (SEQ/64, NHEAD, BSZ), block 128 (4 waves); wave = 16 query rows.
// ---------------------------------------------------------------------------
__global__ void __launch_bounds__(128)
flash_attn(const __bf16* __restrict__ qkv, const __bf16* __restrict__ vT,
           __bf16* __restrict__ attnOut) {
  __shared__ __attribute__((aligned(16))) __bf16 Pl[4][16][80];  // per-wave P tile
  const int lane   = threadIdx.x & 31;
  const int w      = threadIdx.x >> 5;
  const int lane16 = lane & 15;
  const int hsel   = (lane < 16) ? 0 : 1;
  const int qrow0  = blockIdx.x * 64 + w * 16;
  const int h      = blockIdx.y;
  const int b      = blockIdx.z;

  // Q fragments for dh 0..63 (fixed for this wave)
  Frag a0, a1;
  {
    const __bf16* qr = qkv + (size_t)(b * SEQ + qrow0 + lane16) * 3072 + h * HDIM;
    int kb = hsel ? 8 : 0;
    a0.q[0] = *(const u32x4*)(qr + kb);
    a0.q[1] = *(const u32x4*)(qr + kb + 16);
    a1.q[0] = *(const u32x4*)(qr + 32 + kb);
    a1.q[1] = *(const u32x4*)(qr + 32 + kb + 16);
  }

  const f32x8 z = {0.f,0.f,0.f,0.f,0.f,0.f,0.f,0.f};
  f32x8 o[4]; o[0] = z; o[1] = z; o[2] = z; o[3] = z;
  float rmax[8], rsum[8];
#pragma unroll
  for (int c = 0; c < 8; ++c) { rmax[c] = -1e30f; rsum[c] = 0.f; }

  for (int kb0 = 0; kb0 < SEQ; kb0 += 64) {
    // S = Q K^T for a 16x64 tile (4 column sub-tiles)
    f32x8 s[4]; s[0] = z; s[1] = z; s[2] = z; s[3] = z;
    const int dho = hsel ? 8 : 0;
#pragma unroll
    for (int kt = 0; kt < 4; ++kt) {
      const __bf16* kr =
          qkv + (size_t)(b * SEQ + kb0 + kt * 16 + lane16) * 3072 + 1024 + h * HDIM;
      Frag bk;
      bk.q[0] = *(const u32x4*)(kr + dho);
      bk.q[1] = *(const u32x4*)(kr + dho + 16);
      s[kt] = wmma_bf16(a0, bk, s[kt]);
      bk.q[0] = *(const u32x4*)(kr + 32 + dho);
      bk.q[1] = *(const u32x4*)(kr + 32 + dho + 16);
      s[kt] = wmma_bf16(a1, bk, s[kt]);
    }
#pragma unroll
    for (int kt = 0; kt < 4; ++kt) s[kt] *= 0.125f;   // 1/sqrt(64)

    // online softmax: rows align with accumulator components
#pragma unroll
    for (int c = 0; c < 8; ++c) {
      float m = fmaxf(fmaxf(s[0][c], s[1][c]), fmaxf(s[2][c], s[3][c]));
#pragma unroll
      for (int msk = 8; msk >= 1; msk >>= 1) m = fmaxf(m, __shfl_xor(m, msk));
      float nm   = fmaxf(rmax[c], m);
      float corr = __expf(rmax[c] - nm);
      rmax[c] = nm;
      float ts = 0.f;
#pragma unroll
      for (int kt = 0; kt < 4; ++kt) {
        float p = __expf(s[kt][c] - nm);
        s[kt][c] = p;
        ts += p;
      }
#pragma unroll
      for (int msk = 8; msk >= 1; msk >>= 1) ts += __shfl_xor(ts, msk);
      rsum[c] = rsum[c] * corr + ts;
      o[0][c] *= corr; o[1][c] *= corr; o[2][c] *= corr; o[3][c] *= corr;
    }

    // stage P (C-layout) to LDS as row-major bf16, reload as A-fragments
#pragma unroll
    for (int kt = 0; kt < 4; ++kt)
#pragma unroll
      for (int c = 0; c < 8; ++c)
        Pl[w][c + hsel * 8][kt * 16 + lane16] = (__bf16)s[kt][c];
    __syncthreads();

    Frag pa0, pa1;
    {
      int kb = hsel ? 8 : 0;
      pa0.q[0] = *(const u32x4*)&Pl[w][lane16][kb];
      pa0.q[1] = *(const u32x4*)&Pl[w][lane16][kb + 16];
      pa1.q[0] = *(const u32x4*)&Pl[w][lane16][32 + kb];
      pa1.q[1] = *(const u32x4*)&Pl[w][lane16][32 + kb + 16];
    }
    const int ko = hsel ? 8 : 0;
#pragma unroll
    for (int f = 0; f < 4; ++f) {
      const __bf16* vr =
          vT + (size_t)((b * NHEAD + h) * HDIM + f * 16 + lane16) * SEQ;
      Frag bv;
      bv.q[0] = *(const u32x4*)(vr + kb0 + ko);
      bv.q[1] = *(const u32x4*)(vr + kb0 + ko + 16);
      o[f] = wmma_bf16(pa0, bv, o[f]);
      bv.q[0] = *(const u32x4*)(vr + kb0 + 32 + ko);
      bv.q[1] = *(const u32x4*)(vr + kb0 + 32 + ko + 16);
      o[f] = wmma_bf16(pa1, bv, o[f]);
    }
    __syncthreads();
  }

  // normalize + write bf16 [token, h*64+dh]
#pragma unroll
  for (int c = 0; c < 8; ++c) {
    float inv = 1.0f / rsum[c];
    int m = b * SEQ + qrow0 + c + hsel * 8;
#pragma unroll
    for (int f = 0; f < 4; ++f)
      attnOut[(size_t)m * DH + h * HDIM + f * 16 + lane16] = (__bf16)(o[f][c] * inv);
  }
}

// ---------------------------------------------------------------------------
// copy time_emb into the tail of d_out (tuple output: (x, time_emb))
// ---------------------------------------------------------------------------
__global__ void __launch_bounds__(256) copy_temb(const float* __restrict__ temb,
                                                 float* __restrict__ out) {
  int i = blockIdx.x * 256 + threadIdx.x;
  out[(size_t)NTOK * DH + i] = temb[i];
}

// ---------------------------------------------------------------------------
// launcher
// ---------------------------------------------------------------------------
extern "C" void kernel_launch(void* const* d_in, const int* in_sizes, int n_in,
                              void* d_out, int out_size, void* d_ws, size_t ws_size,
                              hipStream_t stream) {
  (void)in_sizes; (void)n_in; (void)out_size; (void)ws_size;
  const float* x      = (const float*)d_in[0];
  const float* temb   = (const float*)d_in[1];
  const float* ln1_g  = (const float*)d_in[2];
  const float* ln1_b  = (const float*)d_in[3];
  const float* s1_w   = (const float*)d_in[4];
  const float* s1_b   = (const float*)d_in[5];
  const float* o1_w   = (const float*)d_in[6];
  const float* o1_b   = (const float*)d_in[7];
  const float* qkv_w  = (const float*)d_in[8];
  const float* qkv_b  = (const float*)d_in[9];
  const float* proj_w = (const float*)d_in[10];
  const float* proj_b = (const float*)d_in[11];
  const float* ln2_g  = (const float*)d_in[12];
  const float* ln2_b  = (const float*)d_in[13];
  const float* s2_w   = (const float*)d_in[14];
  const float* s2_b   = (const float*)d_in[15];
  const float* o2_w   = (const float*)d_in[16];
  const float* o2_b   = (const float*)d_in[17];
  const float* up_w   = (const float*)d_in[18];
  const float* up_b   = (const float*)d_in[19];
  const float* down_w = (const float*)d_in[20];
  const float* down_b = (const float*)d_in[21];

  char* ws = (char*)d_ws;
  // workspace layout (bytes, lifetime-overlapped; peak ~88 MB)
  const size_t o_wqkv  = 0;                                 // 3072*1024 bf16
  const size_t o_wproj = o_wqkv  + (size_t)3072 * 1024 * 2; // 1024*1024 bf16
  const size_t o_wup   = o_wproj + (size_t)1024 * 1024 * 2; // 4096*1024 bf16
  const size_t o_wdown = o_wup   + (size_t)4096 * 1024 * 2; // 1024*4096 bf16
  const size_t o_sc1   = o_wdown + (size_t)1024 * 4096 * 2;
  const size_t o_of1   = o_sc1 + 16384;
  const size_t o_sc2   = o_of1 + 16384;
  const size_t o_of2   = o_sc2 + 16384;
  const size_t o_x1    = o_of2 + 16384;                     // 4096*1024 f32
  const size_t o_hmod  = o_x1   + (size_t)NTOK * DH * 4;    // bf16 (reused as h2)
  const size_t o_qkvb  = o_hmod + (size_t)NTOK * DH * 2;    // 4096*3072 bf16 (reused as upbuf)
  const size_t o_vT    = o_qkvb + (size_t)NTOK * 3072 * 2;  // 4M bf16
  const size_t o_attnb = o_vT   + (size_t)4 * 1024 * 1024 * 2;

  __bf16* wqkv  = (__bf16*)(ws + o_wqkv);
  __bf16* wproj = (__bf16*)(ws + o_wproj);
  __bf16* wup   = (__bf16*)(ws + o_wup);
  __bf16* wdown = (__bf16*)(ws + o_wdown);
  float*  sc1   = (float*)(ws + o_sc1);
  float*  of1   = (float*)(ws + o_of1);
  float*  sc2   = (float*)(ws + o_sc2);
  float*  of2   = (float*)(ws + o_of2);
  float*  x1    = (float*)(ws + o_x1);
  __bf16* hmod  = (__bf16*)(ws + o_hmod);   // LN1 out, later LN2 out
  __bf16* qkvb  = (__bf16*)(ws + o_qkvb);   // later reused as up-proj output
  __bf16* vT    = (__bf16*)(ws + o_vT);
  __bf16* attnb = (__bf16*)(ws + o_attnb);
  __bf16* upbuf = qkvb;                     // 32MB alias over qkvb+vT region

  // 1) weights -> bf16
  f32_to_bf16<<<(3072 * 1024) / 256, 256, 0, stream>>>(qkv_w,  wqkv,  3072 * 1024);
  f32_to_bf16<<<(1024 * 1024) / 256, 256, 0, stream>>>(proj_w, wproj, 1024 * 1024);
  f32_to_bf16<<<(4096 * 1024) / 256, 256, 0, stream>>>(up_w,   wup,   4096 * 1024);
  f32_to_bf16<<<(1024 * 4096) / 256, 256, 0, stream>>>(down_w, wdown, 1024 * 4096);

  // 2) adaLN scale/offset GEMVs (tiny)
  ada_gemv<<<dim3(4, BSZ), 256, 0, stream>>>(temb, s1_w, s1_b, sc1);
  ada_gemv<<<dim3(4, BSZ), 256, 0, stream>>>(temb, o1_w, o1_b, of1);
  ada_gemv<<<dim3(4, BSZ), 256, 0, stream>>>(temb, s2_w, s2_b, sc2);
  ada_gemv<<<dim3(4, BSZ), 256, 0, stream>>>(temb, o2_w, o2_b, of2);

  // 3) LN1 + modulation -> hmod (bf16)
  ln_ada<<<NTOK, 256, 0, stream>>>(x, ln1_g, ln1_b, sc1, of1, hmod);

  // 4) QKV GEMM -> qkvb (bf16, 4096x3072)
  gemm_wmma<false, false, true><<<dim3(3072 / 64, NTOK / 128), 256, 0, stream>>>(
      hmod, wqkv, qkv_b, nullptr, nullptr, qkvb, NTOK, 3072, DH);

  // 5) V transpose for contiguous B-fragments in P*V
  v_transpose<<<(4 * 1024 * 1024) / 256, 256, 0, stream>>>(qkvb, vT);

  // 6) flash attention -> attnb (bf16)
  flash_attn<<<dim3(SEQ / 64, NHEAD, BSZ), 128, 0, stream>>>(qkvb, vT, attnb);

  // 7) proj GEMM + residual(x) -> x1 (f32)
  gemm_wmma<true, false, false><<<dim3(DH / 64, NTOK / 128), 256, 0, stream>>>(
      attnb, wproj, proj_b, x, x1, nullptr, NTOK, DH, DH);

  // 8) LN2 + modulation -> hmod (reused, bf16)
  ln_ada<<<NTOK, 256, 0, stream>>>(x1, ln2_g, ln2_b, sc2, of2, hmod);

  // 9) up GEMM + exact GELU -> upbuf (bf16, 4096x4096)
  gemm_wmma<false, true, true><<<dim3(4096 / 64, NTOK / 128), 256, 0, stream>>>(
      hmod, wup, up_b, nullptr, nullptr, upbuf, NTOK, 4096, DH);

  // 10) down GEMM + residual(x1) -> d_out (f32)
  gemm_wmma<true, false, false><<<dim3(DH / 64, NTOK / 128), 256, 0, stream>>>(
      upbuf, wdown, down_b, x1, (float*)d_out, nullptr, NTOK, DH, 4096);

  // 11) second tuple output: time_emb passthrough
  copy_temb<<<(BSZ * DH) / 256, 256, 0, stream>>>(temb, (float*)d_out);
}
